// CaduceusEncoder_48473000902759
// MI455X (gfx1250) — matrix operand order repaired and checked by reading, not compiled, not run on hardware
//
#include <hip/hip_runtime.h>
#include <hip/hip_bf16.h>

// ---------------------------------------------------------------------------
// CaduceusEncoder for MI455X (gfx1250, wave32, WMMA).
// Both encoder passes (tokens + reverse-complement) are stacked into batch 64.
// GEMMs use v_wmma_f32_16x16x32_bf16; the sequential GRU recurrence runs in 8
// persistent workgroups (2 directions x 4 batch chunks) with the hidden state
// resident in LDS and the gate nonlinearity fused in registers.
// ---------------------------------------------------------------------------

typedef __attribute__((ext_vector_type(16))) __bf16 v16bf;
typedef __attribute__((ext_vector_type(8)))  __bf16 v8bf;
typedef __attribute__((ext_vector_type(8)))  float  v8f;

#define S_LEN 1024
#define DMODEL 256
#define NBATCH 64            // 2 passes * B=32

__device__ inline v8f wmma_bf16(v16bf a, v16bf b, v8f c) {
  return __builtin_amdgcn_wmma_f32_16x16x32_bf16(false, a, false, b, (short)0, c,
                                                 false, false);
}

// A fragment (16x32 bf16, row-major source, stride in elements).
// ISA layout: lanes 0-15 hold row M=lane, K {0..7} U {16..23};
//             lanes 16-31 hold row M=lane-16, K {8..15} U {24..31}.
__device__ inline v16bf load_a_frag(const __bf16* tile, int stride, int lane) {
  int r = lane & 15;
  int sel = lane >> 4;
  const __bf16* p = tile + (size_t)r * stride + sel * 8;
  union { v16bf v; v8bf h[2]; } u;
  u.h[0] = *(const v8bf*)(p);
  u.h[1] = *(const v8bf*)(p + 16);
  return u.v;
}

// B fragment (32x16 bf16) from W stored row-major (N,K): column n of B is
// row n of W. Lanes 0-15: col=lane, K {0..7} U {16..23}; lanes 16-31: col=lane-16, K {8..15} U {24..31}.
__device__ inline v16bf load_b_frag(const __bf16* wtile, int stride, int lane) {
  int n = lane & 15;
  int sel = lane >> 4;
  const __bf16* p = wtile + (size_t)n * stride + sel * 8;
  union { v16bf v; v8bf h[2]; } u;
  u.h[0] = *(const v8bf*)(p);
  u.h[1] = *(const v8bf*)(p + 16);
  return u.v;
}

__device__ inline float sigmoidf_(float x) { return 1.f / (1.f + __expf(-x)); }

// ---------------------------------------------------------------------------
// fp32 -> bf16 conversion
__global__ void cvt_bf16_kernel(const float* __restrict__ in, void* __restrict__ outv,
                                int n) {
  __bf16* out = (__bf16*)outv;
  int i = blockIdx.x * 256 + threadIdx.x;
  if (i < n) out[i] = (__bf16)in[i];
}

// ---------------------------------------------------------------------------
// Embedding + positional; pass 1 uses reverse-complement flipped tokens.
__global__ void embed_kernel(const int* __restrict__ tokens,
                             const float* __restrict__ emb,
                             const float* __restrict__ pos,
                             float* __restrict__ h) {
  size_t gid = (size_t)blockIdx.x * 256 + threadIdx.x;   // NBATCH*S*D threads
  int d = (int)(gid & 255);
  size_t row = gid >> 8;                  // (pass*32+b)*1024 + s
  int s = (int)(row & 1023);
  int pb = (int)(row >> 10);              // 0..63
  int b = pb & 31;
  int pass = pb >> 5;
  int tok;
  if (pass == 0) {
    tok = tokens[b * S_LEN + s];
  } else {
    int t0 = tokens[b * S_LEN + (S_LEN - 1 - s)];
    tok = (t0 == 4) ? 4 : (3 - t0);       // RC_MAP = [3,2,1,0,4]
  }
  h[gid] = emb[tok * DMODEL + d] + pos[s * DMODEL + d];
}

// ---------------------------------------------------------------------------
// LayerNorm over D=256; one wave per row; writes bf16 and/or fp32 outputs.
__global__ __launch_bounds__(256) void ln_kernel(const float* __restrict__ x,
                                                 const float* __restrict__ g,
                                                 const float* __restrict__ b,
                                                 void* __restrict__ outBv,
                                                 float* __restrict__ outF,
                                                 int nrows) {
  __bf16* outB = (__bf16*)outBv;
  int wave = threadIdx.x >> 5;
  int lane = threadIdx.x & 31;
  int row = blockIdx.x * 8 + wave;
  if (row >= nrows) return;
  const float* xr = x + (size_t)row * DMODEL;
  float v[8];
  float s = 0.f;
#pragma unroll
  for (int i = 0; i < 8; ++i) { v[i] = xr[lane + 32 * i]; s += v[i]; }
#pragma unroll
  for (int off = 16; off > 0; off >>= 1) s += __shfl_xor(s, off, 32);
  float mean = s * (1.f / 256.f);
  float q = 0.f;
#pragma unroll
  for (int i = 0; i < 8; ++i) { float d = v[i] - mean; q += d * d; }
#pragma unroll
  for (int off = 16; off > 0; off >>= 1) q += __shfl_xor(q, off, 32);
  float rstd = rsqrtf(q * (1.f / 256.f) + 1e-5f);
#pragma unroll
  for (int i = 0; i < 8; ++i) {
    int col = lane + 32 * i;
    float y = (v[i] - mean) * rstd * g[col] + b[col];
    size_t idx = (size_t)row * DMODEL + col;
    if (outB) outB[idx] = (__bf16)y;
    if (outF) outF[idx] = y;
  }
}

// ---------------------------------------------------------------------------
// WMMA GEMM: C(M,N) = A(M,K)bf16 @ W(N,K)bf16^T + bias, fused epilogues.
// EPI 0: out_bf16 = acc + bias                                     (xi proj)
// EPI 1: g=sigmoid(acc+bias); out_bf16 = g*ex1 + (1-g)*ex2         (gate mix)
// EPI 2: outF = acc + bias + ex1                                   (residual)
// Block: 4 waves, each wave computes a 16x64 strip (4 accumulators).
template <int EPI>
__global__ __launch_bounds__(128) void gemm_kernel(const void* __restrict__ Av,
                                                   const void* __restrict__ Wv,
                                                   const float* __restrict__ bias,
                                                   const float* __restrict__ ex1,
                                                   const float* __restrict__ ex2,
                                                   float* __restrict__ outF,
                                                   void* __restrict__ outBv,
                                                   int M, int N, int K) {
  const __bf16* A = (const __bf16*)Av;
  const __bf16* W = (const __bf16*)Wv;
  __bf16* outB = (__bf16*)outBv;
  const int lane = threadIdx.x & 31;
  const int wave = threadIdx.x >> 5;
  const int m0 = (blockIdx.y * 4 + wave) * 16;
  const int n0 = blockIdx.x * 64;

  const v8f vz = {0.f, 0.f, 0.f, 0.f, 0.f, 0.f, 0.f, 0.f};
  v8f acc[4] = {vz, vz, vz, vz};

  for (int k0 = 0; k0 < K; k0 += 32) {
    v16bf a = load_a_frag(A + (size_t)m0 * K + k0, K, lane);
#pragma unroll
    for (int j = 0; j < 4; ++j) {
      v16bf bfr = load_b_frag(W + (size_t)(n0 + j * 16) * K + k0, K, lane);
      acc[j] = wmma_bf16(a, bfr, acc[j]);
    }
  }

  const int rbase = (lane < 16) ? 0 : 8;
  const int cl = lane & 15;
#pragma unroll
  for (int j = 0; j < 4; ++j) {
    int col = n0 + j * 16 + cl;
    float bv = bias[col];
#pragma unroll
    for (int i = 0; i < 8; ++i) {
      int row = m0 + rbase + i;
      size_t idx = (size_t)row * N + col;
      float v = acc[j][i] + bv;
      if (EPI == 0) {
        outB[idx] = (__bf16)v;
      } else if (EPI == 1) {
        float g = sigmoidf_(v);
        outB[idx] = (__bf16)(g * ex1[idx] + (1.f - g) * ex2[idx]);
      } else {
        outF[idx] = v + ex1[idx];
      }
    }
  }
}

// ---------------------------------------------------------------------------
// Persistent bidirectional GRU recurrence.
// grid = 8 blocks: blockIdx = dir*4 + mchunk (16 batch rows each).
// 16 waves; wave w owns hidden columns [16w,16w+16) across the r/z/n gates so
// the gate nonlinearity is computed entirely in registers.
__global__ __launch_bounds__(512) void gru_rec_kernel(
    const void* __restrict__ xi_fv, const void* __restrict__ xi_rv,
    const void* __restrict__ whh_fv, const void* __restrict__ whh_rv,
    const float* __restrict__ bhh_fv, const float* __restrict__ bhh_rv,
    float* __restrict__ out_f, float* __restrict__ out_r,
    void* __restrict__ fr_bfv) {
  __shared__ float h32[16 * DMODEL];
  __shared__ __bf16 hbf[16 * DMODEL];

  const int tid = threadIdx.x;
  const int lane = tid & 31;
  const int wave = tid >> 5;            // 0..15 -> hidden n-tile
  const int dir = blockIdx.x >> 2;      // 0 = forward, 1 = reverse
  const int m0 = (blockIdx.x & 3) * 16; // global batch-row base

  const __bf16* xi  = (const __bf16*)(dir ? xi_rv : xi_fv);
  const __bf16* whh = (const __bf16*)(dir ? whh_rv : whh_fv);
  const float*  bhh = dir ? bhh_rv : bhh_fv;
  float* outp = dir ? out_r : out_f;
  __bf16* fr_bf = (__bf16*)fr_bfv;

  for (int i = tid; i < 16 * DMODEL; i += 512) {
    h32[i] = 0.f;
    hbf[i] = (__bf16)0.f;
  }
  __syncthreads();

  const int nh0 = wave * 16;
  const int cl = lane & 15;
  const int col = nh0 + cl;
  const float br = bhh[col];
  const float bz = bhh[256 + col];
  const float bn = bhh[512 + col];
  const int rbase = (lane < 16) ? 0 : 8;
  const v8f vz8 = {0.f, 0.f, 0.f, 0.f, 0.f, 0.f, 0.f, 0.f};

  for (int t = 0; t < S_LEN; ++t) {
    const int s = dir ? (S_LEN - 1 - t) : t;

    v8f cr = vz8, cz = vz8, cn = vz8;
    for (int k0 = 0; k0 < DMODEL; k0 += 32) {
      v16bf a = load_a_frag(hbf + k0, DMODEL, lane);
      cr = wmma_bf16(a, load_b_frag(whh + (size_t)(nh0) * DMODEL + k0, DMODEL, lane), cr);
      cz = wmma_bf16(a, load_b_frag(whh + (size_t)(256 + nh0) * DMODEL + k0, DMODEL, lane), cz);
      cn = wmma_bf16(a, load_b_frag(whh + (size_t)(512 + nh0) * DMODEL + k0, DMODEL, lane), cn);
    }
    __syncthreads();  // all A-fragment reads of hbf complete

#pragma unroll
    for (int i = 0; i < 8; ++i) {
      int rl = rbase + i;        // local batch row 0..15
      int b = m0 + rl;           // global batch row 0..63
      size_t xb = ((size_t)b * S_LEN + s) * 768;
      float xr = (float)xi[xb + col];
      float xz = (float)xi[xb + 256 + col];
      float xn = (float)xi[xb + 512 + col];
      float hp = h32[rl * DMODEL + col];
      float r = sigmoidf_(xr + cr[i] + br);
      float z = sigmoidf_(xz + cz[i] + bz);
      float n = tanhf(xn + r * (cn[i] + bn));
      float h2 = (1.f - z) * n + z * hp;
      h32[rl * DMODEL + col] = h2;
      hbf[rl * DMODEL + col] = (__bf16)h2;
      size_t ob = ((size_t)b * S_LEN + s) * DMODEL + col;
      outp[ob] = h2;
      fr_bf[((size_t)b * S_LEN + s) * 512 + dir * 256 + col] = (__bf16)h2;
    }
    __syncthreads();  // h state update visible before next step's GEMM
  }
}

// ---------------------------------------------------------------------------
// pooled[b,d] = 0.5*(mean_s ln(h_pass0) + mean_s ln(h_pass1))
__global__ void pool_kernel(const float* __restrict__ lnf, float* __restrict__ pool) {
  int b = blockIdx.x;
  int d = threadIdx.x;
  float s = 0.f;
  for (int pass = 0; pass < 2; ++pass) {
    const float* base = lnf + ((size_t)(pass * 32 + b) * S_LEN) * DMODEL + d;
    for (int t = 0; t < S_LEN; ++t) s += base[(size_t)t * DMODEL];
  }
  pool[b * DMODEL + d] = s * (1.f / 2048.f);
}

// p = LN(pool@pw.T + pb); out = gelu_exact(p)      (32 blocks x 64 threads)
__global__ __launch_bounds__(64) void head_kernel(const float* __restrict__ pool,
                                                  const float* __restrict__ pw,
                                                  const float* __restrict__ pbias,
                                                  const float* __restrict__ pg,
                                                  const float* __restrict__ pbeta,
                                                  float* __restrict__ out) {
  __shared__ float sp[64];
  int b = blockIdx.x;
  int o = threadIdx.x;
  const float* pr = pool + b * DMODEL;
  const float* wr = pw + o * DMODEL;
  float acc = pbias[o];
  for (int d = 0; d < DMODEL; ++d) acc += pr[d] * wr[d];
  sp[o] = acc;
  __syncthreads();
  float m = 0.f;
  for (int i = 0; i < 64; ++i) m += sp[i];
  m *= (1.f / 64.f);
  float q = 0.f;
  for (int i = 0; i < 64; ++i) { float d = sp[i] - m; q += d * d; }
  q *= (1.f / 64.f);
  float y = (acc - m) * rsqrtf(q + 1e-5f) * pg[o] + pbeta[o];
  out[b * 64 + o] = 0.5f * y * (1.f + erff(y * 0.70710678118654752f));
}

// ---------------------------------------------------------------------------
extern "C" void kernel_launch(void* const* d_in, const int* in_sizes, int n_in,
                              void* d_out, int out_size, void* d_ws, size_t ws_size,
                              hipStream_t stream) {
  const int* tokens  = (const int*)d_in[0];
  const float* emb   = (const float*)d_in[1];
  const float* pos   = (const float*)d_in[2];
  const float* ln_g  = (const float*)d_in[3];
  const float* ln_b  = (const float*)d_in[4];
  const float* wih_f = (const float*)d_in[5];
  const float* whh_f = (const float*)d_in[6];
  const float* bih_f = (const float*)d_in[7];
  const float* bhh_f = (const float*)d_in[8];
  const float* wih_r = (const float*)d_in[9];
  const float* whh_r = (const float*)d_in[10];
  const float* bih_r = (const float*)d_in[11];
  const float* bhh_r = (const float*)d_in[12];
  const float* wg    = (const float*)d_in[13];
  const float* bg    = (const float*)d_in[14];
  const float* wo    = (const float*)d_in[15];
  const float* bo    = (const float*)d_in[16];
  const float* fn_g  = (const float*)d_in[17];
  const float* fn_b  = (const float*)d_in[18];
  const float* pw    = (const float*)d_in[19];
  const float* pb    = (const float*)d_in[20];
  const float* pln_g = (const float*)d_in[21];
  const float* pln_b = (const float*)d_in[22];
  float* out = (float*)d_out;

  char* wsp = (char*)d_ws;
  size_t off = 0;
  auto alloc = [&](size_t bytes) -> char* {
    char* p = wsp + off;
    off += (bytes + 255) & ~(size_t)255;
    return p;
  };
  const size_t MROWS = (size_t)NBATCH * S_LEN;        // 65536
  char* wihf_bf = alloc((size_t)4 * 768 * 256 * 2);
  char* whhf_bf = alloc((size_t)4 * 768 * 256 * 2);
  char* wihr_bf = alloc((size_t)4 * 768 * 256 * 2);
  char* whhr_bf = alloc((size_t)4 * 768 * 256 * 2);
  char* wg_bf   = alloc((size_t)4 * 256 * 512 * 2);
  char* wo_bf   = alloc((size_t)4 * 256 * 256 * 2);
  float* hbuf   = (float*)alloc(MROWS * DMODEL * 4);
  char* hn_bf   = alloc(MROWS * DMODEL * 2);
  char* xi_f    = alloc(MROWS * 768 * 2);
  char* xi_r    = alloc(MROWS * 768 * 2);
  float* f32buf = (float*)alloc(MROWS * DMODEL * 4);  // also reused as final-LN out
  float* r32buf = (float*)alloc(MROWS * DMODEL * 4);
  char* fr_bf   = alloc(MROWS * 512 * 2);
  char* c_bf    = alloc(MROWS * DMODEL * 2);
  float* poolbuf = (float*)alloc(32 * DMODEL * 4);

  // Weight conversion fp32 -> bf16
  {
    int n1 = 4 * 768 * 256;
    cvt_bf16_kernel<<<(n1 + 255) / 256, 256, 0, stream>>>(wih_f, wihf_bf, n1);
    cvt_bf16_kernel<<<(n1 + 255) / 256, 256, 0, stream>>>(whh_f, whhf_bf, n1);
    cvt_bf16_kernel<<<(n1 + 255) / 256, 256, 0, stream>>>(wih_r, wihr_bf, n1);
    cvt_bf16_kernel<<<(n1 + 255) / 256, 256, 0, stream>>>(whh_r, whhr_bf, n1);
    int n2 = 4 * 256 * 512;
    cvt_bf16_kernel<<<(n2 + 255) / 256, 256, 0, stream>>>(wg, wg_bf, n2);
    int n3 = 4 * 256 * 256;
    cvt_bf16_kernel<<<(n3 + 255) / 256, 256, 0, stream>>>(wo, wo_bf, n3);
  }

  // Embedding (both passes stacked into batch 64)
  embed_kernel<<<(unsigned)(MROWS * DMODEL / 256), 256, 0, stream>>>(tokens, emb, pos, hbuf);

  for (int l = 0; l < 4; ++l) {
    // hn = LN(h)  (bf16 for the GEMMs)
    ln_kernel<<<(unsigned)(MROWS / 8), 256, 0, stream>>>(
        hbuf, ln_g + l * 256, ln_b + l * 256, hn_bf, nullptr, (int)MROWS);

    // Input projections: xi = hn @ wih.T + bih   (M=65536, N=768, K=256)
    gemm_kernel<0><<<dim3(768 / 64, (unsigned)(MROWS / 64)), 128, 0, stream>>>(
        hn_bf, wihf_bf + (size_t)l * 768 * 256 * 2, bih_f + l * 768,
        nullptr, nullptr, nullptr, xi_f, (int)MROWS, 768, 256);
    gemm_kernel<0><<<dim3(768 / 64, (unsigned)(MROWS / 64)), 128, 0, stream>>>(
        hn_bf, wihr_bf + (size_t)l * 768 * 256 * 2, bih_r + l * 768,
        nullptr, nullptr, nullptr, xi_r, (int)MROWS, 768, 256);

    // Bidirectional sequential recurrence (persistent blocks)
    gru_rec_kernel<<<8, 512, 0, stream>>>(
        xi_f, xi_r,
        whhf_bf + (size_t)l * 768 * 256 * 2, whhr_bf + (size_t)l * 768 * 256 * 2,
        bhh_f + l * 768, bhh_r + l * 768, f32buf, r32buf, fr_bf);

    // c = sigmoid([f,r]@wg.T + bg) * f + (1-g)*r   (M=65536, N=256, K=512)
    gemm_kernel<1><<<dim3(256 / 64, (unsigned)(MROWS / 64)), 128, 0, stream>>>(
        fr_bf, wg_bf + (size_t)l * 256 * 512 * 2, bg + l * 256,
        f32buf, r32buf, nullptr, c_bf, (int)MROWS, 256, 512);

    // h = c@wo.T + bo + h   (M=65536, N=256, K=256)
    gemm_kernel<2><<<dim3(256 / 64, (unsigned)(MROWS / 64)), 128, 0, stream>>>(
        c_bf, wo_bf + (size_t)l * 256 * 256 * 2, bo + l * 256,
        hbuf, nullptr, hbuf, nullptr, (int)MROWS, 256, 256);
  }

  // Final LN (fp32 out, reuse f32buf), pooling over passes+sequence, head
  ln_kernel<<<(unsigned)(MROWS / 8), 256, 0, stream>>>(
      hbuf, fn_g, fn_b, nullptr, f32buf, (int)MROWS);
  pool_kernel<<<32, 256, 0, stream>>>(f32buf, poolbuf);
  head_kernel<<<32, 64, 0, stream>>>(poolbuf, pw, pb, pln_g, pln_b, out);
}
